// NeRF_5488968204927
// MI455X (gfx1250) — matrix operand order repaired and checked by reading, not compiled, not run on hardware
//
#include <hip/hip_runtime.h>
#include <hip/hip_bf16.h>

typedef __attribute__((ext_vector_type(16))) _Float16 v16h;
typedef __attribute__((ext_vector_type(8)))  _Float16 v8h;
typedef __attribute__((ext_vector_type(8)))  float    v8f;

#define HIDDEN_DIM 256
#define N_SAMPLES  64
#define LDS_STRIDE 264   // 256 + 8 halves pad -> 528B row, 4-bank skew, 16B aligned

// ws layout (halves):
//   [0      , 65536)  WT2[n][k] = (f16) W2[k][n]
//   [65536  , 131072) WT3[n][k] = (f16) W3[k][n]
//   [131072 , 135168) WT4[n][k] = (f16) W4[k][n] for n<4, zero for n in [4,16)
#define WT4_OFF 131072
#define TRANSPOSE_TOTAL 135168   // = 528 * 256

__global__ void transpose_w_kernel(const float* __restrict__ W2,
                                   const float* __restrict__ W3,
                                   const float* __restrict__ W4,
                                   _Float16* __restrict__ WT) {
  const int idx = blockIdx.x * blockDim.x + threadIdx.x;
  if (idx >= TRANSPOSE_TOTAL) return;
  if (idx < 131072) {
    const int layer = idx >> 16;
    const int e = idx & 65535;
    const int n = e >> 8;
    const int k = e & 255;
    const float* W = layer ? W3 : W2;
    WT[layer * 65536 + n * 256 + k] = (_Float16)W[k * 256 + n];
  } else {
    const int e = idx - 131072;        // 0 .. 4095
    const int n = e >> 8;              // 0 .. 15 (padded N)
    const int k = e & 255;
    const float v = (n < 4) ? W4[k * 4 + n] : 0.0f;
    WT[WT4_OFF + n * 256 + k] = (_Float16)v;
  }
}

// One 256->256 dense layer with bias+ReLU. 8 waves: wave -> (rowTile = w>>1, colHalf = w&1).
// Each wave: 8 accumulator tiles (16x16), K = 256 in 8 steps of 32.
// Per K-step: stage all 8 B fragments into registers (one load clause), then 8 WMMAs
// back-to-back so the scheduler can overlap the next step's loads with this step's math.
__device__ __forceinline__ void dense256_wmma(
    const _Float16* __restrict__ Hin, _Float16* __restrict__ Hout,
    const _Float16* __restrict__ WT, const float* __restrict__ bias,
    int lane, int wave) {
  const int rowTile = wave >> 1;               // 0..3  (16 rows each)
  const int n0base  = (wave & 1) * 128;        // column half
  const int hiHalf  = (lane & 16) ? 1 : 0;
  const int m       = rowTile * 16 + (lane & 15);
  const int kloA    = hiHalf ? 16 : 0;         // A: K {0..15} / {16..31} (contiguous)
  const int kloB    = hiHalf ? 16 : 0;         // B: K {0..15} / {16..31} (contiguous)
  const int ncol    = lane & 15;

  v8f acc[8];
  const v8f vzero = {0.f, 0.f, 0.f, 0.f, 0.f, 0.f, 0.f, 0.f};
#pragma unroll
  for (int ct = 0; ct < 8; ++ct) acc[ct] = vzero;

  const _Float16* wbase = WT + n0base * HIDDEN_DIM + ncol * HIDDEN_DIM + kloB;

#pragma unroll
  for (int kk = 0; kk < HIDDEN_DIM; kk += 32) {
    const _Float16* arow = Hin + m * LDS_STRIDE + kk + kloA;
    v8h alo = *reinterpret_cast<const v8h*>(arow);
    v8h ahi = *reinterpret_cast<const v8h*>(arow + 8);
    v16h afrag = __builtin_shufflevector(alo, ahi,
        0, 1, 2, 3, 4, 5, 6, 7, 8, 9, 10, 11, 12, 13, 14, 15);

    v16h bfrag[8];
#pragma unroll
    for (int ct = 0; ct < 8; ++ct)
      bfrag[ct] = *reinterpret_cast<const v16h*>(wbase + ct * 16 * HIDDEN_DIM + kk);

#pragma unroll
    for (int ct = 0; ct < 8; ++ct)
      acc[ct] = __builtin_amdgcn_wmma_f32_16x16x32_f16(
          false, afrag, false, bfrag[ct], (short)0, acc[ct], false, false);
  }

  // Epilogue: bias + ReLU, f32 -> f16, back to LDS.
  // C/D layout: VGPR r -> M = r (lanes 0-15) or r+8 (lanes 16-31); N = n0 + (lane&15).
#pragma unroll
  for (int ct = 0; ct < 8; ++ct) {
    const int n = n0base + ct * 16 + ncol;
    const float bv = bias[n];
#pragma unroll
    for (int r = 0; r < 8; ++r) {
      const int mrow = rowTile * 16 + r + (hiHalf ? 8 : 0);
      Hout[mrow * LDS_STRIDE + n] = (_Float16)fmaxf(acc[ct][r] + bv, 0.0f);
    }
  }
}

__global__ __launch_bounds__(256)
void nerf_fused_kernel(const float* __restrict__ origins,
                       const float* __restrict__ dirs,
                       const float* __restrict__ W1,
                       const float* __restrict__ b1,
                       const float* __restrict__ b2,
                       const float* __restrict__ b3,
                       const float* __restrict__ b4,
                       const _Float16* __restrict__ WT,   // WT2 | WT3 | WT4(padded)
                       const int* __restrict__ nearp,
                       const int* __restrict__ farp,
                       float* __restrict__ out) {
  __shared__ __align__(16) _Float16 Ha[N_SAMPLES * LDS_STRIDE];
  __shared__ __align__(16) _Float16 Hb[N_SAMPLES * LDS_STRIDE];
  __shared__ float raw[N_SAMPLES * 4];     // [s][c] : rgb logits + density logit
  __shared__ float alphaS[N_SAMPLES];

  const int t    = threadIdx.x;
  const int lane = t & 31;
  const int wave = t >> 5;
  const int ray  = blockIdx.x;

  const float nearf = (float)(*nearp);
  const float farf  = (float)(*farp);
  const float dz    = (farf - nearf) * (1.0f / N_SAMPLES);  // uniform segment length

  // ---- Layer 1: points -> Ha (thread t owns hidden column j = t, loops samples) ----
  {
    const float ox = origins[ray * 3 + 0], oy = origins[ray * 3 + 1], oz = origins[ray * 3 + 2];
    const float rx = dirs[ray * 3 + 0],    ry = dirs[ray * 3 + 1],    rz = dirs[ray * 3 + 2];
    const int j = t;
    const float w0 = W1[j], w1 = W1[256 + j], w2 = W1[512 + j], bb = b1[j];
#pragma unroll 4
    for (int s = 0; s < N_SAMPLES; ++s) {
      const float zm = nearf + dz * ((float)s + 0.5f);     // midpoint sample depth
      const float h = fmaf(ox + rx * zm, w0,
                      fmaf(oy + ry * zm, w1,
                      fmaf(oz + rz * zm, w2, bb)));
      Ha[s * LDS_STRIDE + j] = (_Float16)fmaxf(h, 0.0f);
    }
  }
  __syncthreads();

  // ---- Layers 2 & 3: WMMA f16 -> f32 accum ----
  dense256_wmma(Ha, Hb, WT, b2, lane, wave);
  __syncthreads();
  dense256_wmma(Hb, Ha, WT + 65536, b3, lane, wave);
  __syncthreads();

  // ---- Layer 4: 256 -> 4 on the matrix core (W4^T zero-padded to 16 rows) ----
  // Waves 0..3 each own one 16-row tile; one 16x16 C tile, cols 0..3 valid.
  if (wave < 4) {
    const int rowTile = wave;
    const int hiHalf  = (lane & 16) ? 1 : 0;
    const int m       = rowTile * 16 + (lane & 15);
    const int kloA    = hiHalf ? 16 : 0;
    const int kloB    = hiHalf ? 16 : 0;
    const int ncol    = lane & 15;

    v8f acc = {0.f, 0.f, 0.f, 0.f, 0.f, 0.f, 0.f, 0.f};
    const _Float16* w4base = WT + WT4_OFF + ncol * HIDDEN_DIM + kloB;
#pragma unroll
    for (int kk = 0; kk < HIDDEN_DIM; kk += 32) {
      const _Float16* arow = Ha + m * LDS_STRIDE + kk + kloA;
      v8h alo = *reinterpret_cast<const v8h*>(arow);
      v8h ahi = *reinterpret_cast<const v8h*>(arow + 8);
      v16h afrag = __builtin_shufflevector(alo, ahi,
          0, 1, 2, 3, 4, 5, 6, 7, 8, 9, 10, 11, 12, 13, 14, 15);
      v16h bfrag = *reinterpret_cast<const v16h*>(w4base + kk);
      acc = __builtin_amdgcn_wmma_f32_16x16x32_f16(
          false, afrag, false, bfrag, (short)0, acc, false, false);
    }
    if (ncol < 4) {
      const float bv = b4[ncol];
#pragma unroll
      for (int r = 0; r < 8; ++r) {
        const int s = rowTile * 16 + r + (hiHalf ? 8 : 0);
        raw[s * 4 + ncol] = acc[r] + bv;
      }
    }
  }
  __syncthreads();

  // ---- Alpha from density ----
  if (t < N_SAMPLES) {
    const float d = fmaxf(raw[t * 4 + 3], 0.0f);
    alphaS[t] = 1.0f - __expf(-d * dz);
  }
  __syncthreads();

  // ---- Deterministic serial compositing scan: one lane per RGB channel ----
  if (t < 3) {
    float T = 1.0f, accum = 0.0f;
    for (int s = 0; s < N_SAMPLES; ++s) {
      const float a   = alphaS[s];
      const float col = 1.0f / (1.0f + __expf(-raw[s * 4 + t]));  // sigmoid
      accum = fmaf(T * a, col, accum);
      T *= (1.0f - a + 1e-10f);
    }
    out[ray * 3 + t] = accum;
  }
}

extern "C" void kernel_launch(void* const* d_in, const int* in_sizes, int n_in,
                              void* d_out, int out_size, void* d_ws, size_t ws_size,
                              hipStream_t stream) {
  const float* origins = (const float*)d_in[0];
  const float* dirs    = (const float*)d_in[1];
  const float* W1      = (const float*)d_in[2];
  const float* b1      = (const float*)d_in[3];
  const float* W2      = (const float*)d_in[4];
  const float* b2      = (const float*)d_in[5];
  const float* W3      = (const float*)d_in[6];
  const float* b3      = (const float*)d_in[7];
  const float* W4      = (const float*)d_in[8];
  const float* b4      = (const float*)d_in[9];
  const int*   nearp   = (const int*)d_in[10];
  const int*   farp    = (const int*)d_in[11];
  float* out = (float*)d_out;

  _Float16* WT = (_Float16*)d_ws;              // ~264 KB scratch

  const int B = in_sizes[0] / 3;               // 16384 rays

  // One-shot weight transpose+convert (deterministic every call).
  transpose_w_kernel<<<(TRANSPOSE_TOTAL + 255) / 256, 256, 0, stream>>>(W2, W3, W4, WT);

  // Fused MLP + volume rendering: one workgroup (8 waves) per ray.
  nerf_fused_kernel<<<B, 256, 0, stream>>>(origins, dirs, W1, b1, b2, b3,
                                           b4, WT, nearp, farp, out);
}
